// GoalEncoder_13683765805293
// MI455X (gfx1250) — matrix operand order repaired and checked by reading, not compiled, not run on hardware
//
#include <hip/hip_runtime.h>
#include <hip/hip_bf16.h>

typedef __attribute__((ext_vector_type(16))) __bf16 v16bf;
typedef __attribute__((ext_vector_type(8)))  __bf16 v8bf;
typedef __attribute__((ext_vector_type(8)))  float  v8f;

#define T_STEPS 20
#define HID     128
#define G4      512          // 4*HID gate columns
#define EMB_IN  64

// ---- d_ws layout (bytes) ----
#define WS_WC_OFF    0       // 512*2 f32  = 4096 B   (folded W_ih@W_emb)
#define WS_BIAS_OFF  4096    // 512   f32  = 2048 B   (b_ih+b_hh+W_ih@b_emb)
#define WS_WHH_OFF   8192    // 131072 B  bf16 B-fragments of W_hh

// ---- LDS layout (bytes) ----
#define L_WC_OFF     0
#define L_BIAS_OFF   4096
#define L_WHH_OFF    6144                    // 131072 B
#define L_STAGE_OFF  (6144 + 131072)
#define STAGE_ROW    136                     // bf16 units: 128 + 8 pad (272 B, conflict-free transpose)
#define STAGE_WAVE   (16 * STAGE_ROW)        // 2176 bf16 per wave
#define SMEM_BYTES   (L_STAGE_OFF + 4 * STAGE_WAVE * 2)   // 154624 < 320 KB/WGP

__device__ __forceinline__ float sigm(float x) {
    return __builtin_amdgcn_rcpf(1.0f + __expf(-x));
}
__device__ __forceinline__ float tanh_fast(float x) {
    return 2.0f * __builtin_amdgcn_rcpf(1.0f + __expf(-2.0f * x)) - 1.0f;
}

// -------- setup: fold embedding into rank-2 update, swizzle W_hh into bf16 B-fragments --------
__global__ void goalenc_setup_kernel(const float* __restrict__ W_emb, const float* __restrict__ b_emb,
                                     const float* __restrict__ W_ih, const float* __restrict__ W_hh,
                                     const float* __restrict__ b_ih, const float* __restrict__ b_hh,
                                     float* __restrict__ ws_wc, float* __restrict__ ws_bias,
                                     __bf16* __restrict__ ws_whh) {
    const int tid = threadIdx.x;   // 256 threads, 1 block
    for (int col = tid; col < G4; col += 256) {
        float wc0 = 0.f, wc1 = 0.f, bx = 0.f;
        for (int k = 0; k < EMB_IN; ++k) {
            float w = W_ih[col * EMB_IN + k];
            wc0 += w * W_emb[k * 2 + 0];
            wc1 += w * W_emb[k * 2 + 1];
            bx  += w * b_emb[k];
        }
        ws_wc[col * 2 + 0] = wc0;
        ws_wc[col * 2 + 1] = wc1;
        ws_bias[col] = b_ih[col] + b_hh[col] + bx;
    }
    // B-fragment order (ISA 7.12.2): fragment(jg,kc) = 32x16 tile, lane n=l&15 / half=l>>4,
// chunk c elem s holds K = kc*32 + 16*c + 8*half + s. Stored as 2 chunks of 16B per lane, lane-major.
    // idx bits: c(1) | lane(5) | kc(2) | jg(5)  -> 8192 16-byte chunks
    for (int idx = tid; idx < 8192; idx += 256) {
        const int c    = idx & 1;
        const int lane = (idx >> 1) & 31;
        const int kc   = (idx >> 6) & 3;
        const int jg   = idx >> 8;
        const int n    = lane & 15, half = lane >> 4;
        const int col  = jg * 16 + n;
        __bf16* dst = ws_whh + ((((jg * 4 + kc) * 1024) + c * 512 + lane * 16) >> 1);
        for (int s = 0; s < 8; ++s) {
            const int k = kc * 32 + c * 16 + half * 8 + s;
            dst[s] = (__bf16)W_hh[col * HID + k];
        }
    }
}

// -------- main: 4 waves/block, 16 batch rows/wave, h/c resident for all 20 steps --------
__global__ __launch_bounds__(128) void goalenc_lstm_wmma_kernel(
    const float* __restrict__ obs, const float* __restrict__ h0, const float* __restrict__ c0,
    const float* __restrict__ ws_wc, const float* __restrict__ ws_bias,
    const __bf16* __restrict__ ws_whh, float* __restrict__ out, int batch) {
    extern __shared__ char smem[];
    float*  s_wc    = (float*)(smem + L_WC_OFF);
    float*  s_bias  = (float*)(smem + L_BIAS_OFF);
    char*   s_whh   = smem + L_WHH_OFF;
    __bf16* s_stage = (__bf16*)(smem + L_STAGE_OFF);

    const int tid = threadIdx.x;
    // cooperative LDS fill (weights shared by all 4 waves, loaded once per block)
    for (int i = tid; i < G4 * 2; i += 128) s_wc[i]   = ws_wc[i];
    for (int i = tid; i < G4;     i += 128) s_bias[i] = ws_bias[i];
    {
        const uint4* src = (const uint4*)ws_whh;
        uint4*       dst = (uint4*)s_whh;
        for (int i = tid; i < 8192; i += 128) dst[i] = src[i];
    }
    __syncthreads();

    const int wave = tid >> 5, lane = tid & 31;
    const int half = lane >> 4, n = lane & 15;
    const int rowbase = blockIdx.x * 64 + wave * 16;
    __bf16* stage = s_stage + wave * STAGE_WAVE;

    // ---- init c (f32, C/D tile layout) and h0 -> bf16 staging ----
    v8f cst[8];
#pragma unroll
    for (int jt = 0; jt < 8; ++jt) {
#pragma unroll
        for (int r = 0; r < 8; ++r) {
            const int m = r + 8 * half;
            const size_t gi = (size_t)(rowbase + m) * HID + jt * 16 + n;
            cst[jt][r] = c0[gi];
            stage[m * STAGE_ROW + jt * 16 + n] = (__bf16)h0[gi];
        }
    }

    // ---- build A-fragments of h (ISA 7.12.2 A layout: lane m=l&15; elems 0-7 -> K=half*8+0..7,
    //      elems 8-15 -> K=16+half*8+0..7), i.e. two 16B LDS chunks per kc ----
    v16bf A[4];
#pragma unroll
    for (int kc = 0; kc < 4; ++kc) {
        const char* base = (const char*)stage + (lane & 15) * (STAGE_ROW * 2) + kc * 64 + half * 16;
        v8bf lo = *(const v8bf*)base;
        v8bf hi = *(const v8bf*)(base + 32);
        A[kc] = __builtin_shufflevector(lo, hi, 0,1,2,3,4,5,6,7,8,9,10,11,12,13,14,15);
    }

#pragma unroll 1
    for (int t = 0; t < T_STEPS; ++t) {
        // obs for this wave's 16 rows (broadcast across lane halves)
        float o0[8], o1[8];
#pragma unroll
        for (int r = 0; r < 8; ++r) {
            const int m = r + 8 * half;
            const float* p = obs + ((size_t)t * batch + rowbase + m) * 2;
            o0[r] = p[0];
            o1[r] = p[1];
        }
#pragma unroll
        for (int jt = 0; jt < 8; ++jt) {
            v8f acc[4];
            // rank-2 obs update + bias seeds the accumulators (x-path GEMM folded away)
#pragma unroll
            for (int G = 0; G < 4; ++G) {
                const int col = (G * 8 + jt) * 16 + n;
                const float wc0 = s_wc[col * 2 + 0], wc1 = s_wc[col * 2 + 1], bb = s_bias[col];
#pragma unroll
                for (int r = 0; r < 8; ++r)
                    acc[G][r] = fmaf(o0[r], wc0, fmaf(o1[r], wc1, bb));
            }
            // h @ W_hh^T over K=128: 4 kc-chunks x 4 gates = 16 WMMAs per jt-tile
#pragma unroll
            for (int kc = 0; kc < 4; ++kc) {
#pragma unroll
                for (int G = 0; G < 4; ++G) {
                    const int jg = G * 8 + jt;
                    const char* fb = s_whh + ((jg * 4 + kc) << 10) + lane * 16;
                    v8bf b0 = *(const v8bf*)fb;
                    v8bf b1 = *(const v8bf*)(fb + 512);
                    v16bf B = __builtin_shufflevector(b0, b1, 0,1,2,3,4,5,6,7,8,9,10,11,12,13,14,15);
                    acc[G] = __builtin_amdgcn_wmma_f32_16x16x32_bf16(
                        false, A[kc], false, B, (short)0, acc[G], false, false);
                }
            }
            // gate nonlinearities (TRANS ops co-execute with XDL WMMA), c in registers
#pragma unroll
            for (int r = 0; r < 8; ++r) {
                const float iv = sigm(acc[0][r]);
                const float fv = sigm(acc[1][r]);
                const float gv = tanh_fast(acc[2][r]);
                const float ov = sigm(acc[3][r]);
                const float cv = fv * cst[jt][r] + iv * gv;
                cst[jt][r] = cv;
                const float hv = ov * tanh_fast(cv);
                const int m = r + 8 * half;
                stage[m * STAGE_ROW + jt * 16 + n] = (__bf16)hv;
                if (t == T_STEPS - 1)   // wave-uniform branch: EXEC stays all-1s
                    out[(size_t)(rowbase + m) * HID + jt * 16 + n] = hv;
            }
        }
        // rebuild A-fragments for next step from per-wave staging (same-wave LDS: in-order)
#pragma unroll
        for (int kc = 0; kc < 4; ++kc) {
            const char* base = (const char*)stage + (lane & 15) * (STAGE_ROW * 2) + kc * 64 + half * 16;
            v8bf lo = *(const v8bf*)base;
            v8bf hi = *(const v8bf*)(base + 32);
            A[kc] = __builtin_shufflevector(lo, hi, 0,1,2,3,4,5,6,7,8,9,10,11,12,13,14,15);
        }
    }
}

extern "C" void kernel_launch(void* const* d_in, const int* in_sizes, int n_in,
                              void* d_out, int out_size, void* d_ws, size_t ws_size,
                              hipStream_t stream) {
    const float* obs   = (const float*)d_in[0];
    const float* h0    = (const float*)d_in[1];
    const float* c0    = (const float*)d_in[2];
    const float* W_emb = (const float*)d_in[3];
    const float* b_emb = (const float*)d_in[4];
    const float* W_ih  = (const float*)d_in[5];
    const float* W_hh  = (const float*)d_in[6];
    const float* b_ih  = (const float*)d_in[7];
    const float* b_hh  = (const float*)d_in[8];
    float* out = (float*)d_out;

    char*   ws      = (char*)d_ws;
    float*  ws_wc   = (float*)(ws + WS_WC_OFF);
    float*  ws_bias = (float*)(ws + WS_BIAS_OFF);
    __bf16* ws_whh  = (__bf16*)(ws + WS_WHH_OFF);

    const int batch = in_sizes[1] / HID;   // h0 is [B, HID]

    hipLaunchKernelGGL(goalenc_setup_kernel, dim3(1), dim3(256), 0, stream,
                       W_emb, b_emb, W_ih, W_hh, b_ih, b_hh, ws_wc, ws_bias, ws_whh);

    hipFuncSetAttribute(reinterpret_cast<const void*>(goalenc_lstm_wmma_kernel),
                        hipFuncAttributeMaxDynamicSharedMemorySize, SMEM_BYTES);
    hipLaunchKernelGGL(goalenc_lstm_wmma_kernel, dim3(batch / 64), dim3(128), SMEM_BYTES, stream,
                       obs, h0, c0, ws_wc, ws_bias, ws_whh, out, batch);
}
